// get_model_28956669510153
// MI455X (gfx1250) — matrix operand order repaired and checked by reading, not compiled, not run on hardware
//
#include <hip/hip_runtime.h>
#include <hip/hip_bf16.h>

// ---------------------------------------------------------------------------
// PointNet++ classification pipeline for MI455X (gfx1250, wave32, WMMA).
// All matmuls: bf16 WMMA GEMM, 64x64 block tile, 4 waves, 4x
// v_wmma_f32_16x16x32_bf16 per wave per K-step. LDS staged with gfx1250
// global_load_async_to_lds_b128, double-buffered (wait asynccnt<=4) so the
// next K-tile streams in while the current one is in the matrix pipe.
// ---------------------------------------------------------------------------

typedef unsigned short bf16_t;
typedef __attribute__((ext_vector_type(16))) __bf16 v16bf;
typedef __attribute__((ext_vector_type(8)))  float  v8f;
typedef int v4i_gcc __attribute__((vector_size(16)));

#define BATCH 16
#define NPTS  8192

__device__ __forceinline__ bf16_t f2bf(float f) {
    union { float f; unsigned u; } x; x.f = f;
    return (bf16_t)((x.u + 0x7FFFu + ((x.u >> 16) & 1u)) >> 16);
}
__device__ __forceinline__ float bf2f(bf16_t b) {
    union { unsigned u; float f; } x; x.u = ((unsigned)b) << 16;
    return x.f;
}

// ---- 16-byte global->LDS tile copy; async path when available --------------
#if __has_builtin(__builtin_amdgcn_global_load_async_to_lds_b128)
#define TILE_ASYNC 1
#else
#define TILE_ASYNC 0
#endif

__device__ __forceinline__ void tile_cp16(const void* g, void* l) {
#if TILE_ASYNC
    __builtin_amdgcn_global_load_async_to_lds_b128(
        (__attribute__((address_space(1))) v4i_gcc*)g,
        (__attribute__((address_space(3))) v4i_gcc*)l, 0, 0);
#else
    *(uint4*)l = *(const uint4*)g;
#endif
}
__device__ __forceinline__ void tile_cp_wait0() {
#if TILE_ASYNC
#if __has_builtin(__builtin_amdgcn_s_wait_asynccnt)
    __builtin_amdgcn_s_wait_asynccnt(0);
#else
    asm volatile("s_wait_asynccnt 0x0" ::: "memory");
#endif
#endif
}
__device__ __forceinline__ void tile_cp_wait4() {
#if TILE_ASYNC
#if __has_builtin(__builtin_amdgcn_s_wait_asynccnt)
    __builtin_amdgcn_s_wait_asynccnt(4);
#else
    asm volatile("s_wait_asynccnt 0x4" ::: "memory");
#endif
#endif
}

// --------------------------- split [B,6,N] -> coords/norms [B,N,3] ----------
__global__ void split_kernel(const float* __restrict__ X, float* __restrict__ coords,
                             float* __restrict__ norms, int N) {
    size_t i = (size_t)blockIdx.x * blockDim.x + threadIdx.x;
    size_t total = (size_t)BATCH * N;
    if (i >= total) return;
    int b = (int)(i / N), j = (int)(i % N);
    const float* src = X + (size_t)b * 6 * N;
    for (int c = 0; c < 3; ++c) {
        coords[i * 3 + c] = src[c * N + j];
        norms[i * 3 + c]  = src[(3 + c) * N + j];
    }
}

// --------------------------- farthest point sampling ------------------------
__global__ void fps_kernel(const float* __restrict__ pts, int N, int npoint,
                           int* __restrict__ idx, float* __restrict__ new_xyz) {
    __shared__ float dist[NPTS];
    __shared__ float rv[256];
    __shared__ int   ri[256];
    __shared__ int   s_far;
    const int b = blockIdx.x;
    const int t = threadIdx.x;
    const float* P = pts + (size_t)b * N * 3;
    for (int j = t; j < N; j += blockDim.x) dist[j] = 1e10f;
    if (t == 0) s_far = 0;
    __syncthreads();
    for (int it = 0; it < npoint; ++it) {
        int far = s_far;
        if (t == 0) {
            idx[b * npoint + it] = far;
            new_xyz[((size_t)b * npoint + it) * 3 + 0] = P[far * 3 + 0];
            new_xyz[((size_t)b * npoint + it) * 3 + 1] = P[far * 3 + 1];
            new_xyz[((size_t)b * npoint + it) * 3 + 2] = P[far * 3 + 2];
        }
        float cx = P[far * 3 + 0], cy = P[far * 3 + 1], cz = P[far * 3 + 2];
        float bestd = -1.f; int besti = 0;
        for (int j = t; j < N; j += blockDim.x) {
            float dx = P[j * 3 + 0] - cx;
            float dy = P[j * 3 + 1] - cy;
            float dz = P[j * 3 + 2] - cz;
            float d  = dx * dx + dy * dy + dz * dz;
            float od = dist[j];
            float nd = od < d ? od : d;
            dist[j] = nd;
            if (nd > bestd) { bestd = nd; besti = j; }
        }
        rv[t] = bestd; ri[t] = besti;
        __syncthreads();
        for (int s = blockDim.x / 2; s > 0; s >>= 1) {
            if (t < s) {
                if (rv[t + s] > rv[t] ||
                    (rv[t + s] == rv[t] && ri[t + s] < ri[t])) {
                    rv[t] = rv[t + s]; ri[t] = ri[t + s];
                }
            }
            __syncthreads();
        }
        if (t == 0) s_far = ri[0];
        __syncthreads();
    }
}

// --------------------------- ball query + group -----------------------------
__global__ void group_kernel(const float* __restrict__ xyz,
                             const float* __restrict__ featsF,
                             const bf16_t* __restrict__ featsB,
                             const float* __restrict__ new_xyz,
                             bf16_t* __restrict__ grouped,
                             int N, int S, int nsample, int D, int Kp, float r2) {
    const int bs = blockIdx.x;
    const int b  = bs / S;
    const int lane = threadIdx.x;
    __shared__ int sidx[64];
    __shared__ int s_cnt;
    const float* P = xyz + (size_t)b * N * 3;
    float cx = new_xyz[(size_t)bs * 3 + 0];
    float cy = new_xyz[(size_t)bs * 3 + 1];
    float cz = new_xyz[(size_t)bs * 3 + 2];
    if (lane == 0) s_cnt = 0;
    __syncthreads();
    for (int base = 0; base < N; base += 32) {
        int j = base + lane;
        bool within = false;
        if (j < N) {
            float dx = P[j * 3 + 0] - cx;
            float dy = P[j * 3 + 1] - cy;
            float dz = P[j * 3 + 2] - cz;
            within = (dx * dx + dy * dy + dz * dz) <= r2;
        }
        unsigned mask = (unsigned)__ballot(within);
        int cnt = s_cnt;
        int pre = __popc(mask & ((1u << lane) - 1u));
        if (within && (cnt + pre) < nsample) sidx[cnt + pre] = j;
        __syncthreads();
        if (lane == 0) s_cnt = cnt + __popc(mask);
        __syncthreads();
        if (s_cnt >= nsample) break;
    }
    int have = s_cnt;
    if (have == 0 && lane == 0) sidx[0] = 0;
    __syncthreads();
    int cnt = have == 0 ? 1 : (have < nsample ? have : nsample);
    int first = sidx[0];
    __syncthreads();
    for (int j = lane; j < nsample; j += 32)
        if (j >= cnt) sidx[j] = first;
    __syncthreads();
    bf16_t* G = grouped + (size_t)bs * nsample * Kp;
    for (int jj = 0; jj < nsample; ++jj) {
        int pj = sidx[jj];
        bf16_t* Gr = G + (size_t)jj * Kp;
        for (int c = lane; c < Kp; c += 32) {
            float v;
            if (c < 3) {
                float cv = (c == 0) ? cx : (c == 1) ? cy : cz;
                v = P[pj * 3 + c] - cv;
            } else if (c < 3 + D) {
                size_t fo = ((size_t)b * N + pj) * D + (c - 3);
                v = featsB ? bf2f(featsB[fo]) : featsF[fo];
            } else {
                v = 0.f;
            }
            Gr[c] = f2bf(v);
        }
    }
}

// --------------------------- weight pack fp32 -> bf16 (K padded) ------------
__global__ void packw_kernel(const float* __restrict__ W, bf16_t* __restrict__ Wb,
                             int Cout, int K, int Kp) {
    size_t i = (size_t)blockIdx.x * blockDim.x + threadIdx.x;
    size_t total = (size_t)Cout * Kp;
    if (i >= total) return;
    int r = (int)(i / Kp), c = (int)(i % Kp);
    Wb[i] = (c < K) ? f2bf(W[(size_t)r * K + c]) : (bf16_t)0;
}

// --------------------------- fp32 -> bf16 convert ---------------------------
__global__ void tobf_kernel(const float* __restrict__ X, bf16_t* __restrict__ Y, size_t n) {
    size_t i = (size_t)blockIdx.x * blockDim.x + threadIdx.x;
    if (i < n) Y[i] = f2bf(X[i]);
}

// --------------------------- WMMA bf16 GEMM ---------------------------------
// Y = epilogue( X[M,Kp](bf16) @ W[Cout,Kp](bf16)^T )
// block = 128 threads (4 waves), block tile 64x64, wave tile 16x64.
// ALIGNED=true: M%64==0 && Cout%64==0 -> no bounds checks, double-buffered
// async staging overlapped with WMMA compute.
// mode 0: fp32 out, y = acc + bias
// mode 1: bf16 out, y = relu((acc+bias)*inv*g + beta)
template <bool ALIGNED>
__global__ __launch_bounds__(128)
void gemm_wmma(const bf16_t* __restrict__ X, const bf16_t* __restrict__ W,
               const float* __restrict__ bias, const float* __restrict__ gamma,
               const float* __restrict__ beta, void* __restrict__ Yv,
               int M, int Kp, int Cout, int Ystride, int mode) {
    __shared__ uint4 aT[2][64 * 4];   // ping-pong: 64 rows x 32 bf16 each
    __shared__ uint4 bT[2][64 * 4];
    const int t    = threadIdx.x;
    const int lane = t & 31;
    const int wv   = t >> 5;
    const int half = lane >> 4;
    const int l15  = lane & 15;
    const int m0   = blockIdx.x * 64;
    const int n0   = blockIdx.y * 64;

    v8f acc[4] = {};
    union Frag { uint4 q[2]; v16bf v; };

    // stage one 64x32 A tile + 64x32 B tile into buffer `buf`:
    // 4 async b128 ops per thread (2 A + 2 B)
    auto stage = [&](int buf, int k0) {
        #pragma unroll
        for (int qq = 0; qq < 2; ++qq) {
            int q = t + qq * 128;
            int r = q >> 2, s = q & 3;
            int gm = m0 + r;
            int gn = n0 + r;
            if (ALIGNED || gm < M)
                tile_cp16(X + (size_t)gm * Kp + k0 + s * 8, &aT[buf][q]);
            else
                aT[buf][q] = make_uint4(0, 0, 0, 0);
            if (ALIGNED || gn < Cout)
                tile_cp16(W + (size_t)gn * Kp + k0 + s * 8, &bT[buf][q]);
            else
                bT[buf][q] = make_uint4(0, 0, 0, 0);
        }
    };

    const int nsteps = Kp >> 5;
    stage(0, 0);
    for (int s0 = 0; s0 < nsteps; ++s0) {
        const int cur = s0 & 1;
        if (ALIGNED) {
            if (s0 + 1 < nsteps) {
                stage(cur ^ 1, (s0 + 1) * 32);  // prefetch next tile (async)
                tile_cp_wait4();                // current tile's 4 ops retired
            } else {
                tile_cp_wait0();
            }
        } else {
            tile_cp_wait0();
        }
        __syncthreads();

        // A fragment: lane half 0 -> K{0..7,16..23}, half 1 -> K{8..15,24..31}
        Frag fa;
        const int arow = wv * 16 + l15;
        fa.q[0] = aT[cur][arow * 4 + half];
        fa.q[1] = aT[cur][arow * 4 + half + 2];
        #pragma unroll
        for (int nt = 0; nt < 4; ++nt) {
            // B fragment: column = l15, half 0 -> K{0..7,16..23}, half 1 -> K{8..15,24..31}
            Frag fb;
            const int brow = nt * 16 + l15;
            fb.q[0] = bT[cur][brow * 4 + half];
            fb.q[1] = bT[cur][brow * 4 + half + 2];
            acc[nt] = __builtin_amdgcn_wmma_f32_16x16x32_bf16(
                          false, fa.v, false, fb.v, (short)0, acc[nt], false, false);
        }
        __syncthreads();

        if (!ALIGNED && s0 + 1 < nsteps)
            stage((s0 + 1) & 1, (s0 + 1) * 32);
    }

    // epilogue: C/D layout -> VGPR j holds M = j + 8*half, N = l15
    const float inv = 0.9999950000374997f;  // 1/sqrt(1+1e-5)
    #pragma unroll
    for (int nt = 0; nt < 4; ++nt) {
        int gn = n0 + nt * 16 + l15;
        if (!ALIGNED && gn >= Cout) continue;
        float bb = bias ? bias[gn] : 0.f;
        float gg = (mode != 0) ? gamma[gn] : 1.f;
        float be = (mode != 0) ? beta[gn] : 0.f;
        #pragma unroll
        for (int j = 0; j < 8; ++j) {
            int gm = m0 + wv * 16 + half * 8 + j;
            if (!ALIGNED && gm >= M) continue;
            float v = acc[nt][j] + bb;
            if (mode != 0) {
                v = v * inv * gg + be;
                v = v > 0.f ? v : 0.f;
            }
            if (mode == 1)
                ((bf16_t*)Yv)[(size_t)gm * Ystride + gn] = f2bf(v);
            else
                ((float*)Yv)[(size_t)gm * Ystride + gn] = v;
        }
    }
}

// --------------------------- max pool over samples (bf16 in) ----------------
__global__ void maxpool_kernel(const bf16_t* __restrict__ X, void* __restrict__ Yv,
                               int ns, int C, int out_fp32) {
    int g = blockIdx.x;
    for (int c = threadIdx.x; c < C; c += blockDim.x) {
        const bf16_t* p = X + (size_t)g * ns * C + c;
        float m = -3.4e38f;
        for (int j = 0; j < ns; ++j) {
            float v = bf2f(p[(size_t)j * C]);
            m = v > m ? v : m;
        }
        if (out_fp32) ((float*)Yv)[(size_t)g * C + c] = m;
        else          ((bf16_t*)Yv)[(size_t)g * C + c] = f2bf(m);
    }
}

// --------------------------- concat [rows,3]f32 ++ [rows,D]bf16 -> bf16 Kp --
__global__ void concat_kernel(const float* __restrict__ A, const bf16_t* __restrict__ Bf,
                              bf16_t* __restrict__ O, int rows, int D, int Kp) {
    size_t i = (size_t)blockIdx.x * blockDim.x + threadIdx.x;
    size_t total = (size_t)rows * Kp;
    if (i >= total) return;
    int r = (int)(i / Kp);
    int c = (int)(i % Kp);
    bf16_t v;
    if (c < 3)          v = f2bf(A[(size_t)r * 3 + c]);
    else if (c < 3 + D) v = Bf[(size_t)r * D + (c - 3)];
    else                v = 0;
    O[i] = v;
}

// --------------------------- log-softmax ------------------------------------
__global__ void logsoftmax_kernel(const float* __restrict__ H, float* __restrict__ O, int C) {
    int b = blockIdx.x;
    if (threadIdx.x != 0) return;
    const float* h = H + (size_t)b * C;
    float mx = -3.4e38f;
    for (int c = 0; c < C; ++c) mx = h[c] > mx ? h[c] : mx;
    float s = 0.f;
    for (int c = 0; c < C; ++c) s += __expf(h[c] - mx);
    float ls = __logf(s);
    for (int c = 0; c < C; ++c) O[(size_t)b * C + c] = h[c] - mx - ls;
}

// ---------------------------------------------------------------------------
extern "C" void kernel_launch(void* const* d_in, const int* in_sizes, int n_in,
                              void* d_out, int out_size, void* d_ws, size_t ws_size,
                              hipStream_t stream) {
    const int B = BATCH, N = NPTS;
    const float* xyz = (const float*)d_in[0];

    const float* saW[9]; const float* saB[9]; const float* saG[9]; const float* saBe[9];
    for (int l = 0; l < 9; ++l) {
        saW[l]  = (const float*)d_in[1 + l * 4 + 0];
        saB[l]  = (const float*)d_in[1 + l * 4 + 1];
        saG[l]  = (const float*)d_in[1 + l * 4 + 2];
        saBe[l] = (const float*)d_in[1 + l * 4 + 3];
    }
    const float* W1  = (const float*)d_in[37];
    const float* b1  = (const float*)d_in[38];
    const float* g1  = (const float*)d_in[39];
    const float* be1 = (const float*)d_in[40];
    const float* W2  = (const float*)d_in[41];
    const float* b2  = (const float*)d_in[42];
    const float* g2  = (const float*)d_in[43];
    const float* be2 = (const float*)d_in[44];
    const float* W3  = (const float*)d_in[45];
    const float* b3  = (const float*)d_in[46];

    // byte allocator, 64B aligned (needed for uint4 tile copies)
    char* wp = (char*)d_ws;
    auto allocB = [&](size_t bytes) {
        char* p = wp; wp += (bytes + 63) & ~(size_t)63; return (void*)p;
    };
    float* coords   = (float*)allocB((size_t)B * N * 3 * 4);
    float* norms    = (float*)allocB((size_t)B * N * 3 * 4);
    float* newxyz1  = (float*)allocB((size_t)B * 512 * 3 * 4);
    int*   idx1     = (int*)allocB((size_t)B * 512 * 4);
    float* newxyz2  = (float*)allocB((size_t)B * 128 * 3 * 4);
    int*   idx2     = (int*)allocB((size_t)B * 128 * 4);
    float* h3       = (float*)allocB((size_t)B * 40 * 4);

    const size_t M1 = (size_t)B * 512 * 32;   // 262144
    const size_t M2 = (size_t)B * 128 * 64;   // 131072
    const size_t M3 = (size_t)B * 128;        // 2048

    bf16_t* grouped1 = (bf16_t*)allocB(M1 * 32 * 2);
    bf16_t* grouped2 = (bf16_t*)allocB(M2 * 160 * 2);
    bf16_t* grouped3 = (bf16_t*)allocB(M3 * 288 * 2);
    bf16_t* bufA     = (bf16_t*)allocB(M1 * 128 * 2);
    bf16_t* bufB     = (bf16_t*)allocB(M1 * 128 * 2);
    bf16_t* l1p      = (bf16_t*)allocB((size_t)B * 512 * 128 * 2);
    bf16_t* l2p      = (bf16_t*)allocB((size_t)B * 128 * 256 * 2);
    bf16_t* l3bf     = (bf16_t*)allocB((size_t)B * 1024 * 2);
    bf16_t* fc1      = (bf16_t*)allocB((size_t)B * 512 * 2);
    bf16_t* fc2      = (bf16_t*)allocB((size_t)B * 256 * 2);

    // packed weights (bf16, K padded to mult of 32)
    const int wK [12] = {6, 64, 64, 131, 128, 128, 259, 256, 512, 1024, 512, 256};
    const int wKp[12] = {32, 64, 64, 160, 128, 128, 288, 256, 512, 1024, 512, 256};
    const int wCo[12] = {64, 64, 128, 128, 128, 256, 256, 512, 1024, 512, 256, 40};
    const float* wSrc[12] = {saW[0], saW[1], saW[2], saW[3], saW[4], saW[5],
                             saW[6], saW[7], saW[8], W1, W2, W3};
    bf16_t* Wb[12];
    for (int i = 0; i < 12; ++i) {
        Wb[i] = (bf16_t*)allocB((size_t)wCo[i] * wKp[i] * 2);
        size_t total = (size_t)wCo[i] * wKp[i];
        packw_kernel<<<(int)((total + 255) / 256), 256, 0, stream>>>(
            wSrc[i], Wb[i], wCo[i], wK[i], wKp[i]);
    }

    float* out_logp = (float*)d_out;
    float* out_l3   = (float*)d_out + B * 40;

    auto gemm = [&](const bf16_t* X, int wi, const float* bb, const float* gg,
                    const float* be, void* Y, int M, int Cout, int Ystride, int mode) {
        dim3 grid((M + 63) / 64, (Cout + 63) / 64);
        if ((M % 64 == 0) && (Cout % 64 == 0))
            gemm_wmma<true><<<grid, 128, 0, stream>>>(X, Wb[wi], bb, gg, be, Y,
                                                      M, wKp[wi], Cout, Ystride, mode);
        else
            gemm_wmma<false><<<grid, 128, 0, stream>>>(X, Wb[wi], bb, gg, be, Y,
                                                       M, wKp[wi], Cout, Ystride, mode);
    };

    {   size_t total = (size_t)B * N;
        split_kernel<<<(int)((total + 255) / 256), 256, 0, stream>>>(xyz, coords, norms, N); }

    // ---- SA1: 8192 -> 512, r=0.2, ns=32, [6,64,64,128] ----
    fps_kernel<<<B, 256, 0, stream>>>(coords, N, 512, idx1, newxyz1);
    group_kernel<<<B * 512, 32, 0, stream>>>(coords, norms, nullptr, newxyz1, grouped1,
                                             N, 512, 32, 3, 32, 0.2f * 0.2f);
    gemm(grouped1, 0, saB[0], saG[0], saBe[0], bufA, (int)M1, 64, 64, 1);
    gemm(bufA,     1, saB[1], saG[1], saBe[1], bufB, (int)M1, 64, 64, 1);
    gemm(bufB,     2, saB[2], saG[2], saBe[2], bufA, (int)M1, 128, 128, 1);
    maxpool_kernel<<<B * 512, 128, 0, stream>>>(bufA, l1p, 32, 128, 0);

    // ---- SA2: 512 -> 128, r=0.4, ns=64, [131,128,128,256] ----
    fps_kernel<<<B, 256, 0, stream>>>(newxyz1, 512, 128, idx2, newxyz2);
    group_kernel<<<B * 128, 32, 0, stream>>>(newxyz1, nullptr, l1p, newxyz2, grouped2,
                                             512, 128, 64, 128, 160, 0.4f * 0.4f);
    gemm(grouped2, 3, saB[3], saG[3], saBe[3], bufA, (int)M2, 128, 128, 1);
    gemm(bufA,     4, saB[4], saG[4], saBe[4], bufB, (int)M2, 128, 128, 1);
    gemm(bufB,     5, saB[5], saG[5], saBe[5], bufA, (int)M2, 256, 256, 1);
    maxpool_kernel<<<B * 128, 256, 0, stream>>>(bufA, l2p, 64, 256, 0);

    // ---- SA3: group_all over 128 pts, [259,256,512,1024] ----
    {   size_t total = M3 * 288;
        concat_kernel<<<(int)((total + 255) / 256), 256, 0, stream>>>(
            newxyz2, l2p, grouped3, (int)M3, 256, 288); }
    gemm(grouped3, 6, saB[6], saG[6], saBe[6], bufA, (int)M3, 256, 256, 1);
    gemm(bufA,     7, saB[7], saG[7], saBe[7], bufB, (int)M3, 512, 512, 1);
    gemm(bufB,     8, saB[8], saG[8], saBe[8], bufA, (int)M3, 1024, 1024, 1);
    maxpool_kernel<<<B, 256, 0, stream>>>(bufA, out_l3, 128, 1024, 1);  // l3 fp32 [16,1024]

    // ---- FC head ----
    {   size_t total = (size_t)B * 1024;
        tobf_kernel<<<(int)((total + 255) / 256), 256, 0, stream>>>(out_l3, l3bf, total); }
    gemm(l3bf, 9,  b1, g1, be1, fc1, B, 512, 512, 1);
    gemm(fc1,  10, b2, g2, be2, fc2, B, 256, 256, 1);
    gemm(fc2,  11, b3, nullptr, nullptr, h3, B, 40, 40, 0);
    logsoftmax_kernel<<<B, 1, 0, stream>>>(h3, out_logp, 40);
}
